// OptimizedSelfAttention_67808943670064
// MI455X (gfx1250) — matrix operand order repaired and checked by reading, not compiled, not run on hardware
//
#include <hip/hip_runtime.h>
#include <hip/hip_bf16.h>

typedef __attribute__((ext_vector_type(16))) __bf16 v16bf;
typedef __attribute__((ext_vector_type(8)))  __bf16 v8bf;
typedef __attribute__((ext_vector_type(8)))  float  v8f;

#define Bsz 2
#define Ssz 2048
#define Dsz 1024
#define Hn  16
#define HD  64

__device__ __forceinline__ v8f v8f_zero() {
  v8f z = {0.f,0.f,0.f,0.f,0.f,0.f,0.f,0.f};
  return z;
}

__device__ __forceinline__ v16bf frag_cat(v8bf lo, v8bf hi) {
  return __builtin_shufflevector(lo, hi, 0,1,2,3,4,5,6,7,8,9,10,11,12,13,14,15);
}

// A fragment (16xK=32 bf16): row M = lane%16.
// lane<16 -> K {0..7, 16..23}; lane>=16 -> K {8..15, 24..31}  (ISA 7.12.2)
__device__ __forceinline__ v16bf load_frag_a(const __bf16* base, int stride) {
  const int lane = threadIdx.x & 31;
  const int half = lane >> 4;
  const int r    = lane & 15;
  const __bf16* p = base + r * stride;
  v8bf lo = *(const v8bf*)(p + half * 8);
  v8bf hi = *(const v8bf*)(p + 16 + half * 8);
  return frag_cat(lo, hi);
}

// B fragment (K=32 x 16 bf16) from an [N][K] row-major tile:
// col N = lane%16, lane<16 -> K {0..7, 16..23}; lane>=16 -> K {8..15, 24..31}
__device__ __forceinline__ v16bf load_frag_b(const __bf16* baseT, int stride) {
  const int lane = threadIdx.x & 31;
  const int kh   = lane >> 4;
  const int c    = lane & 15;
  const __bf16* p = baseT + c * stride;
  v8bf lo = *(const v8bf*)(p + kh * 8);
  v8bf hi = *(const v8bf*)(p + 16 + kh * 8);
  return frag_cat(lo, hi);
}

__device__ __forceinline__ v8f wmma_bf16(v16bf a, v16bf b, v8f c) {
  return __builtin_amdgcn_wmma_f32_16x16x32_bf16(false, a, false, b, (short)0, c,
                                                 false, false);
}

__device__ __forceinline__ v8bf cvt8(float4 a, float4 b) {
  v8bf v;
  v[0] = (__bf16)a.x; v[1] = (__bf16)a.y; v[2] = (__bf16)a.z; v[3] = (__bf16)a.w;
  v[4] = (__bf16)b.x; v[5] = (__bf16)b.y; v[6] = (__bf16)b.z; v[7] = (__bf16)b.w;
  return v;
}

// Async DMA: global (64-bit VA) -> LDS (32-bit offset), 16 bytes/lane.
// Tracked on ASYNCcnt (ISA 15.18.3 op 98); bypasses VGPRs entirely.
__device__ __forceinline__ void async_load_b128(unsigned lds_off, const __bf16* g) {
  asm volatile("global_load_async_to_lds_b128 %0, %1, off"
               :: "v"(lds_off), "v"(g)
               : "memory");
}

__device__ __forceinline__ unsigned lds_off_of(const __bf16* p) {
  // generic pointers to LDS carry the LDS byte offset in the low 32 bits
  return (unsigned)(unsigned long long)p;
}

// ---------------------------------------------------------------------------
// Kernel 1: qkv = x @ w_qkv^T  (f32 -> bf16 WMMA), scatter to Q*scale, K, V^T
// M = B*S = 4096, N = 3D = 3072, K = 1024.  Block: 256 thr (8 waves),
// tile 128(M) x 64(N).  Register-pipelined: global loads for step t+1 are in
// flight while step t's WMMAs run (s_wait_loadcnt lands at next ds_store).
// ---------------------------------------------------------------------------
__global__ __launch_bounds__(256) void qkv_gemm(const float* __restrict__ x,
                                                const float* __restrict__ w_qkv,
                                                __bf16* __restrict__ q_ws,
                                                __bf16* __restrict__ k_ws,
                                                __bf16* __restrict__ vt_ws) {
  __shared__ __bf16 At[128 * 40];
  __shared__ __bf16 Bt[64 * 40];
  const int gm0 = blockIdx.y * 128;
  const int gn0 = blockIdx.x * 64;
  const int t = threadIdx.x;
  const int wave = t >> 5, lane = t & 31;

  const int arow = t >> 1, ac0 = (t & 1) * 16;   // A tile: 2 thr/row, 16 elems each
  const int bn   = t >> 2, bc0 = (t & 3) * 8;    // B tile: 4 thr/row,  8 elems each
  const float* aptr = x     + (size_t)(gm0 + arow) * Dsz + ac0;
  const float* bptr = w_qkv + (size_t)(gn0 + bn)   * Dsz + bc0;

  v8f acc[4];
#pragma unroll
  for (int nf = 0; nf < 4; ++nf) acc[nf] = v8f_zero();

  float4 fa0, fa1, fa2, fa3, fb0, fb1;
  {  // preload K-step 0
    const float4* sa = (const float4*)aptr;
    fa0 = sa[0]; fa1 = sa[1]; fa2 = sa[2]; fa3 = sa[3];
    const float4* sb = (const float4*)bptr;
    fb0 = sb[0]; fb1 = sb[1];
  }

  for (int kk = 0; kk < Dsz; kk += 32) {
    // drain in-flight loads into LDS (cvt f32->bf16)
    *(v8bf*)&At[arow * 40 + ac0]     = cvt8(fa0, fa1);
    *(v8bf*)&At[arow * 40 + ac0 + 8] = cvt8(fa2, fa3);
    *(v8bf*)&Bt[bn * 40 + bc0]       = cvt8(fb0, fb1);
    __syncthreads();

    // issue next step's global loads (wraps harmlessly on the last step)
    const int kn = (kk + 32 < Dsz) ? kk + 32 : 0;
    {
      const float4* sa = (const float4*)(aptr + kn);
      fa0 = sa[0]; fa1 = sa[1]; fa2 = sa[2]; fa3 = sa[3];
      const float4* sb = (const float4*)(bptr + kn);
      fb0 = sb[0]; fb1 = sb[1];
    }

    v16bf a = load_frag_a(&At[(wave * 16) * 40], 40);
#pragma unroll
    for (int nf = 0; nf < 4; ++nf) {
      v16bf b = load_frag_b(&Bt[(nf * 16) * 40], 40);
      acc[nf] = wmma_bf16(a, b, acc[nf]);
    }
    __syncthreads();
  }

  // Epilogue: scatter into Q (scaled, [B,H,S,HD]), K ([B,H,S,HD]), V^T ([B,H,HD,S])
  const int half = lane >> 4, c = lane & 15;
#pragma unroll
  for (int nf = 0; nf < 4; ++nf) {
    const int n = gn0 + nf * 16 + c;
    const int part = n >> 10, inner = n & 1023;
    const int h = inner >> 6, hd = inner & 63;
#pragma unroll
    for (int v = 0; v < 8; ++v) {
      const int m = gm0 + wave * 16 + half * 8 + v;
      const int b = m >> 11, s = m & 2047;
      const int bh = b * Hn + h;
      const float val = acc[nf][v];
      if (part == 0)
        q_ws[((size_t)bh * Ssz + s) * HD + hd] = (__bf16)(val * 0.125f);  // 1/sqrt(64)
      else if (part == 1)
        k_ws[((size_t)bh * Ssz + s) * HD + hd] = (__bf16)val;
      else
        vt_ws[((size_t)bh * HD + hd) * Ssz + s] = (__bf16)val;
    }
  }
}

// ---------------------------------------------------------------------------
// Kernel 2: flash attention per (b,h, 64-query tile).  Block: 128 thr (4 waves).
// K/V tiles are DMA'd global->LDS with global_load_async_to_lds_b128 into a
// double buffer; next tile's DMA overlaps current tile's WMMAs, retired with
// s_wait_asynccnt (4 in-flight ops per wave per tile, in-order completion).
// ---------------------------------------------------------------------------
__global__ __launch_bounds__(128) void flash_attn(const __bf16* __restrict__ q_ws,
                                                  const __bf16* __restrict__ k_ws,
                                                  const __bf16* __restrict__ vt_ws,
                                                  __bf16* __restrict__ attn_ws) {
  __shared__ __bf16 Kt[2][32 * 72];    // [key][hd]  (== B^T layout for QK^T)
  __shared__ __bf16 Vt[2][64 * 40];    // [hd][key]  (== B^T layout for P.V)
  __shared__ __bf16 Pt[4][16 * 40];    // per-wave P scratch, [qrow][key]

  const int bh = blockIdx.x >> 5;             // b*H + h
  const int qb = (blockIdx.x & 31) * 64;
  const int t = threadIdx.x, wave = t >> 5, lane = t & 31;
  const int half = lane >> 4, c = lane & 15;
  const int qrow0 = qb + wave * 16;

  // Q fragments (reused across the whole key loop), loaded straight from global
  const __bf16* Qbase = q_ws + ((size_t)bh * Ssz + qrow0) * HD;
  v16bf aq0 = load_frag_a(Qbase, HD);        // hd 0..31
  v16bf aq1 = load_frag_a(Qbase + 32, HD);   // hd 32..63

  // per-thread DMA slots: 2x b128 for K tile, 2x b128 for V tile
  const int kr = t >> 2, kc0 = (t & 3) * 16;  // K tile 32x64
  const int vr = t >> 1, vc0 = (t & 1) * 16;  // V tile 64x32
  const __bf16* kbase = k_ws  + ((size_t)bh * Ssz + kr) * HD + kc0;  // + kb*HD
  const __bf16* vbase = vt_ws + ((size_t)bh * HD + vr) * Ssz + vc0;  // + kb

  auto issue_tile = [&](int buf, int kb) {
    const unsigned ko = lds_off_of(&Kt[buf][kr * 72 + kc0]);
    async_load_b128(ko,      kbase + (size_t)kb * HD);
    async_load_b128(ko + 16, kbase + (size_t)kb * HD + 8);
    const unsigned vo = lds_off_of(&Vt[buf][vr * 40 + vc0]);
    async_load_b128(vo,      vbase + kb);
    async_load_b128(vo + 16, vbase + kb + 8);
  };

  v8f o[4];
  float mi[8], li[8];
#pragma unroll
  for (int v = 0; v < 8; ++v) { mi[v] = -1e30f; li[v] = 0.f; }
#pragma unroll
  for (int nf = 0; nf < 4; ++nf) o[nf] = v8f_zero();

  issue_tile(0, 0);

  for (int kt = 0; kt < Ssz / 32; ++kt) {
    const int kb = kt * 32;
    // DMA the next tile into the other buffer (wraps harmlessly at the end),
    // then retire the current tile: 8 outstanding, oldest 4 == current tile.
    issue_tile((kt + 1) & 1, (kb + 32) & (Ssz - 1));
    asm volatile("s_wait_asynccnt 0x4" ::: "memory");
    __syncthreads();

    const __bf16* Ktb = &Kt[kt & 1][0];
    const __bf16* Vtb = &Vt[kt & 1][0];

    // scores: two 16x16 fragments over keys kb..kb+15 / kb+16..kb+31
    v8f s0 = v8f_zero(), s1 = v8f_zero();
    {
      v16bf b00 = load_frag_b(Ktb, 72);                // keys 0..15, hd 0..31
      v16bf b01 = load_frag_b(Ktb + 32, 72);           // keys 0..15, hd 32..63
      s0 = wmma_bf16(aq0, b00, s0);
      s0 = wmma_bf16(aq1, b01, s0);
      v16bf b10 = load_frag_b(Ktb + 16 * 72, 72);      // keys 16..31, hd 0..31
      v16bf b11 = load_frag_b(Ktb + 16 * 72 + 32, 72); // keys 16..31, hd 32..63
      s1 = wmma_bf16(aq0, b10, s1);
      s1 = wmma_bf16(aq1, b11, s1);
    }

    // Online softmax: rows of the C/D layout = 8*half + v, spread across the
    // 16 lanes of the half -> reduce with xor-shuffles over masks 1,2,4,8.
    float p0[8], p1[8];
#pragma unroll
    for (int v = 0; v < 8; ++v) {
      float mx = fmaxf(s0[v], s1[v]);
#pragma unroll
      for (int d = 1; d < 16; d <<= 1) mx = fmaxf(mx, __shfl_xor(mx, d, 32));
      const float mnew  = fmaxf(mi[v], mx);
      const float alpha = __expf(mi[v] - mnew);
      const float e0 = __expf(s0[v] - mnew);
      const float e1 = __expf(s1[v] - mnew);
      float rs = e0 + e1;
#pragma unroll
      for (int d = 1; d < 16; d <<= 1) rs += __shfl_xor(rs, d, 32);
      li[v] = alpha * li[v] + rs;
      mi[v] = mnew;
      p0[v] = e0;
      p1[v] = e1;
#pragma unroll
      for (int nf = 0; nf < 4; ++nf) o[nf][v] = o[nf][v] * alpha;
    }

    // C/D layout -> A layout for P via per-wave LDS round trip
#pragma unroll
    for (int v = 0; v < 8; ++v) {
      Pt[wave][(half * 8 + v) * 40 + c]      = (__bf16)p0[v];
      Pt[wave][(half * 8 + v) * 40 + 16 + c] = (__bf16)p1[v];
    }
    asm volatile("s_wait_dscnt 0" ::: "memory");  // same-wave LDS RAW fence
    v16bf ap = load_frag_a(&Pt[wave][0], 40);
#pragma unroll
    for (int nf = 0; nf < 4; ++nf) {
      v16bf bv = load_frag_b(Vtb + (nf * 16) * 40, 40);
      o[nf] = wmma_bf16(ap, bv, o[nf]);
    }
    __syncthreads();  // all waves done reading this buffer before its reuse
  }

  // normalize and store attn [B,S,D] as bf16
  const int bb = bh >> 4, h = bh & 15;
#pragma unroll
  for (int v = 0; v < 8; ++v) {
    const float rcp = 1.f / li[v];
    const int s = qrow0 + half * 8 + v;
#pragma unroll
    for (int nf = 0; nf < 4; ++nf) {
      attn_ws[((size_t)(bb * Ssz + s)) * Dsz + h * HD + nf * 16 + c] =
          (__bf16)(o[nf][v] * rcp);
    }
  }
}

// ---------------------------------------------------------------------------
// Kernel 3: out = attn @ w_out^T  (bf16 WMMA, f32 epilogue).  M=4096,N=K=1024.
// Register-pipelined like kernel 1 (A already bf16, B needs f32->bf16 cvt).
// ---------------------------------------------------------------------------
__global__ __launch_bounds__(256) void out_gemm(const __bf16* __restrict__ attn_ws,
                                                const float* __restrict__ w_out,
                                                float* __restrict__ out) {
  __shared__ __bf16 At[128 * 40];
  __shared__ __bf16 Bt[64 * 40];
  const int gm0 = blockIdx.y * 128;
  const int gn0 = blockIdx.x * 64;
  const int t = threadIdx.x;
  const int wave = t >> 5, lane = t & 31;

  const int arow = t >> 1, ac0 = (t & 1) * 16;
  const int bn   = t >> 2, bc0 = (t & 3) * 8;
  const __bf16* aptr = attn_ws + (size_t)(gm0 + arow) * Dsz + ac0;
  const float*  bptr = w_out   + (size_t)(gn0 + bn)   * Dsz + bc0;

  v8f acc[4];
#pragma unroll
  for (int nf = 0; nf < 4; ++nf) acc[nf] = v8f_zero();

  v8bf ra0, ra1;
  float4 fb0, fb1;
  {
    const v8bf* sa = (const v8bf*)aptr;
    ra0 = sa[0]; ra1 = sa[1];
    const float4* sb = (const float4*)bptr;
    fb0 = sb[0]; fb1 = sb[1];
  }

  for (int kk = 0; kk < Dsz; kk += 32) {
    *(v8bf*)&At[arow * 40 + ac0]     = ra0;
    *(v8bf*)&At[arow * 40 + ac0 + 8] = ra1;
    *(v8bf*)&Bt[bn * 40 + bc0]       = cvt8(fb0, fb1);
    __syncthreads();

    const int kn = (kk + 32 < Dsz) ? kk + 32 : 0;
    {
      const v8bf* sa = (const v8bf*)(aptr + kn);
      ra0 = sa[0]; ra1 = sa[1];
      const float4* sb = (const float4*)(bptr + kn);
      fb0 = sb[0]; fb1 = sb[1];
    }

    v16bf a = load_frag_a(&At[(wave * 16) * 40], 40);
#pragma unroll
    for (int nf = 0; nf < 4; ++nf) {
      v16bf b = load_frag_b(&Bt[(nf * 16) * 40], 40);
      acc[nf] = wmma_bf16(a, b, acc[nf]);
    }
    __syncthreads();
  }

  const int half = lane >> 4, c = lane & 15;
#pragma unroll
  for (int nf = 0; nf < 4; ++nf) {
    const int n = gn0 + nf * 16 + c;
#pragma unroll
    for (int v = 0; v < 8; ++v) {
      const int m = gm0 + wave * 16 + half * 8 + v;
      out[(size_t)m * Dsz + n] = acc[nf][v];
    }
  }
}

extern "C" void kernel_launch(void* const* d_in, const int* in_sizes, int n_in,
                              void* d_out, int out_size, void* d_ws, size_t ws_size,
                              hipStream_t stream) {
  (void)in_sizes; (void)n_in; (void)out_size; (void)ws_size;
  const float* x     = (const float*)d_in[0];
  const float* w_qkv = (const float*)d_in[1];
  const float* w_out = (const float*)d_in[2];
  float* out = (float*)d_out;

  const size_t per = (size_t)Bsz * Hn * Ssz * HD;   // 4M bf16 elements = 8 MB each
  __bf16* q_ws  = (__bf16*)d_ws;
  __bf16* k_ws  = q_ws + per;
  __bf16* vt_ws = k_ws + per;
  __bf16* attn  = vt_ws + per;                      // total 32 MB of workspace

  // M=4096, N=3072 -> grid (3072/64, 4096/128)
  qkv_gemm<<<dim3(48, 32), 256, 0, stream>>>(x, w_qkv, q_ws, k_ws, vt_ws);
  // B*H * (S/64) = 32 * 32 = 1024 blocks
  flash_attn<<<dim3(1024), 128, 0, stream>>>(q_ws, k_ws, vt_ws, attn);
  // M=4096, N=1024 -> grid (1024/64, 4096/128)
  out_gemm<<<dim3(16, 32), 256, 0, stream>>>(attn, w_out, out);
}
